// SVDLlamaBlock_49022756716596
// MI455X (gfx1250) — matrix operand order repaired and checked by reading, not compiled, not run on hardware
//
#include <hip/hip_runtime.h>

// ---------------------------------------------------------------------------
// Types / helpers
// ---------------------------------------------------------------------------
typedef __bf16 bf16;
typedef __attribute__((ext_vector_type(16))) __bf16 v16bf;
typedef __attribute__((ext_vector_type(8))) float v8f;

union Frag16 { v16bf v; uint4 q[2]; };
union Frag8f { v8f v; float e[8]; };

__device__ inline bf16 f2bf(float f) {
  union { float f; unsigned u; } a; a.f = f;
  unsigned r = (a.u + 0x7FFFu + ((a.u >> 16) & 1u)) >> 16;  // RNE
  union { unsigned short s; bf16 b; } o; o.s = (unsigned short)r;
  return o.b;
}
__device__ inline float bf2f(bf16 b) {
  union { unsigned short s; bf16 b; } i; i.b = b;
  union { unsigned u; float f; } o; o.u = ((unsigned)i.s) << 16;
  return o.f;
}

// Async global->LDS copy, 16B per lane (GLOBAL_LOAD_ASYNC_TO_LDS_B128, GVS mode).
// ldsOff: LDS byte offset (low 32 bits of generic __shared__ address);
// gOff:   per-lane byte offset added to the 64-bit SGPR base.
__device__ inline void async_ld_b128(unsigned ldsOff, const void* gbase, unsigned gOff) {
  asm volatile("global_load_async_to_lds_b128 %0, %1, %2"
               :
               : "v"(ldsOff), "v"(gOff), "s"(gbase)
               : "memory");
}
__device__ inline void wait_async0() {
  asm volatile("s_wait_asynccnt 0x0" ::: "memory");
}

// ---------------------------------------------------------------------------
// Generic tiled WMMA GEMM:  C[M,N] = A[M,K](row-major, bf16) * W[N,K]^T (bf16)
// Batched via blockIdx.z with (outer,inner) stride pairs (z/bi, z%bi).
// EPI: 0 = f32 out, 1 = bf16 out, 2 = f32 out + residual R (same indexing as C)
// K-loop is double-buffered with async global->LDS copies (ASYNCcnt), one
// barrier per 32-deep K step; LDS rows padded to 40 bf16 (80B = 5x16B) so the
// b128 fragment reads are bank-conflict free.
// ---------------------------------------------------------------------------
template <int MF, int NF, int WM, int WN, int EPI>
__global__ __launch_bounds__(WM * WN * 32) void gemm_bf16_wmma(
    const bf16* __restrict__ A, const bf16* __restrict__ Bw,
    void* __restrict__ Cout, const float* __restrict__ R,
    int K, int ldA, int ldB, int ldC,
    long long aOut, long long aIn, long long bOut, long long bIn,
    long long cOut, long long cIn, int bi)
{
  constexpr int BM = WM * MF * 16;
  constexpr int BN = WN * NF * 16;
  constexpr int NT = WM * WN * 32;
  constexpr int LS = 40;  // padded LDS stride (elements)
  __shared__ bf16 sA[2][BM * LS];
  __shared__ bf16 sB[2][BN * LS];

  const int tid = threadIdx.x;
  const int lane = tid & 31;
  const int wave = tid >> 5;
  const int wm = wave / WN, wn = wave % WN;
  const int lq = lane & 15;
  const int hi16 = (lane >> 4) & 1;

  const int z = blockIdx.z;
  const long long aOff = (long long)(z / bi) * aOut + (long long)(z % bi) * aIn;
  const long long bOff = (long long)(z / bi) * bOut + (long long)(z % bi) * bIn;
  const long long cOff = (long long)(z / bi) * cOut + (long long)(z % bi) * cIn;

  const int m0 = blockIdx.y * BM;
  const int n0 = blockIdx.x * BN;
  const bf16* Ap = A + aOff + (long long)m0 * ldA;
  const bf16* Bp = Bw + bOff + (long long)n0 * ldB;

  auto stage = [&](int buf, int kb) {
    for (int i = tid; i < BM * 4; i += NT) {
      const int row = i >> 2, col = (i & 3) << 3;
      async_ld_b128((unsigned)(size_t)&sA[buf][row * LS + col], Ap,
                    (unsigned)(((long long)row * ldA + kb + col) * 2));
    }
    for (int i = tid; i < BN * 4; i += NT) {
      const int row = i >> 2, col = (i & 3) << 3;
      async_ld_b128((unsigned)(size_t)&sB[buf][row * LS + col], Bp,
                    (unsigned)(((long long)row * ldB + kb + col) * 2));
    }
  };

  Frag8f acc[MF][NF] = {};
  const int nK = K >> 5;
  stage(0, 0);

  for (int ki = 0; ki < nK; ++ki) {
    const int cur = ki & 1;
    wait_async0();        // my async fills of buffer `cur` are done
    __syncthreads();      // everyone's fills done; everyone done reading buf cur^1
    if (ki + 1 < nK) stage(cur ^ 1, (ki + 1) * 32);

    // A frag (16x32 bf16, ISA 7.12.2): lanes 0-15 K={0..7,16..23}, lanes 16-31 +8
    // B frag (32x16): same K order as A frag
    Frag16 af[MF], bfr[NF];
    const int ka = hi16 ? 8 : 0;
    const int kbq = hi16 ? 8 : 0;
#pragma unroll
    for (int fm = 0; fm < MF; ++fm) {
      const bf16* p = &sA[cur][(wm * MF * 16 + fm * 16 + lq) * LS + ka];
      af[fm].q[0] = *(const uint4*)p;
      af[fm].q[1] = *(const uint4*)(p + 16);
    }
#pragma unroll
    for (int fn = 0; fn < NF; ++fn) {
      const bf16* p = &sB[cur][(wn * NF * 16 + fn * 16 + lq) * LS + kbq];
      bfr[fn].q[0] = *(const uint4*)p;
      bfr[fn].q[1] = *(const uint4*)(p + 16);
    }
#pragma unroll
    for (int fm = 0; fm < MF; ++fm)
#pragma unroll
      for (int fn = 0; fn < NF; ++fn)
        acc[fm][fn].v = __builtin_amdgcn_wmma_f32_16x16x32_bf16(
            false, af[fm].v, false, bfr[fn].v, (short)0, acc[fm][fn].v, false, false);
  }

  // Epilogue; C frag: lanes 0-15 M=r, lanes 16-31 M=r+8, N=lane%16
#pragma unroll
  for (int fm = 0; fm < MF; ++fm)
#pragma unroll
    for (int fn = 0; fn < NF; ++fn) {
      const int n = n0 + wn * NF * 16 + fn * 16 + lq;
#pragma unroll
      for (int r = 0; r < 8; ++r) {
        const int m = m0 + wm * MF * 16 + fm * 16 + r + hi16 * 8;
        const long long idx = cOff + (long long)m * ldC + n;
        if constexpr (EPI == 0) ((float*)Cout)[idx] = acc[fm][fn].e[r];
        else if constexpr (EPI == 1) ((bf16*)Cout)[idx] = f2bf(acc[fm][fn].e[r]);
        else ((float*)Cout)[idx] = acc[fm][fn].e[r] + R[idx];
      }
    }
}

template <int MF, int NF, int WM, int WN, int EPI>
static void launch_gemm(hipStream_t st, const bf16* A, const bf16* Bw, void* C,
                        const float* R, int M, int N, int K,
                        int ldA, int ldB, int ldC,
                        int batch = 1, long long aOut = 0, long long aIn = 0,
                        long long bOut = 0, long long bIn = 0,
                        long long cOut = 0, long long cIn = 0, int bi = 1) {
  constexpr int BM = WM * MF * 16, BN = WN * NF * 16;
  dim3 g(N / BN, M / BM, batch), b(WM * WN * 32);
  gemm_bf16_wmma<MF, NF, WM, WN, EPI><<<g, b, 0, st>>>(
      A, Bw, C, R, K, ldA, ldB, ldC, aOut, aIn, bOut, bIn, cOut, cIn, bi);
}

// ---------------------------------------------------------------------------
// Flash attention, transposed orientation.
// Block: 4 waves x 16 q-rows = 64 q rows of one (b,h). Streams 64-wide s chunks
// staged in LDS by async copies. S^T = K * Q^T keeps softmax row-stats lane-
// local (8 elems + one xor-16 shfl). ctx^T = V^T * P^T accumulated in WMMA;
// P re-laid out through per-wave LDS.
// ---------------------------------------------------------------------------
__global__ __launch_bounds__(128) void attn_flash_wmma(
    const bf16* __restrict__ qr, const bf16* __restrict__ kr,
    const bf16* __restrict__ vt, bf16* __restrict__ ctx,
    int H_, int HK_, int T_)
{
  const int qblk = blockIdx.x;
  const int bh = blockIdx.y;
  const int b = bh / H_, h = bh % H_;
  const int hk = h / (H_ / HK_);
  const int tid = threadIdx.x;
  const int lane = tid & 31;
  const int w = tid >> 5;
  const int lq = lane & 15;
  const int hi16 = (lane >> 4) & 1;

  __shared__ bf16 sK[64 * 72];      // [s][dh], padded
  __shared__ bf16 sVT[64 * 72];     // [dh][s], padded
  __shared__ bf16 sP[4][16 * 72];   // per-wave [q][s]

  const bf16* kbase = kr + ((long long)(b * HK_ + hk) * T_) * 64;
  const bf16* vbase = vt + ((long long)(b * HK_ + hk)) * 64LL * T_;

  const int qrow = qblk * 64 + w * 16 + lq;
  const bf16* qp = qr + ((long long)bh * T_ + qrow) * 64;
  Frag16 qT[2];  // B-frags of Q^T (k=dh, n=q), resident across the s loop
#pragma unroll
  for (int st = 0; st < 2; ++st) {
    const int dh0 = st * 32 + (hi16 ? 8 : 0);
    qT[st].q[0] = *(const uint4*)(qp + dh0);
    qT[st].q[1] = *(const uint4*)(qp + dh0 + 16);
  }

  float m_run = -3.0e38f, l_run = 0.0f;
  Frag8f cacc[4] = {};  // ctx^T: M=dh (4 frags), N=q

  const int nChunks = qblk + 1;  // causal: s0 <= q block end
  for (int c = 0; c < nChunks; ++c) {
    const int s0 = c * 64;
    __syncthreads();   // everyone done reading previous chunk's sK/sVT
    for (int i = tid; i < 512; i += 128) {
      const int row = i >> 3, col = (i & 7) << 3;
      async_ld_b128((unsigned)(size_t)&sK[row * 72 + col], kbase,
                    (unsigned)((((long long)(s0 + row)) * 64 + col) * 2));
      async_ld_b128((unsigned)(size_t)&sVT[row * 72 + col], vbase,
                    (unsigned)(((long long)row * T_ + s0 + col) * 2));
    }
    wait_async0();
    __syncthreads();

    // S^T = K * Q^T
    Frag8f sacc[4] = {};
#pragma unroll
    for (int st = 0; st < 2; ++st) {
      const int koff = st * 32 + (hi16 ? 8 : 0);
#pragma unroll
      for (int fm = 0; fm < 4; ++fm) {
        Frag16 a;
        const bf16* p = &sK[(fm * 16 + lq) * 72 + koff];
        a.q[0] = *(const uint4*)p;
        a.q[1] = *(const uint4*)(p + 16);
        sacc[fm].v = __builtin_amdgcn_wmma_f32_16x16x32_bf16(
            false, a.v, false, qT[st].v, (short)0, sacc[fm].v, false, false);
      }
    }

    // scale + causal mask + online softmax (per-lane; lane pair shares q row)
    float cmax = -3.0e38f;
    float pv[4][8];
#pragma unroll
    for (int fm = 0; fm < 4; ++fm)
#pragma unroll
      for (int r = 0; r < 8; ++r) {
        const int sg = s0 + fm * 16 + r + hi16 * 8;
        float vv = sacc[fm].e[r] * 0.125f;  // 1/sqrt(64)
        if (sg > qrow) vv = -3.0e38f;
        pv[fm][r] = vv;
        cmax = fmaxf(cmax, vv);
      }
    cmax = fmaxf(cmax, __shfl_xor(cmax, 16));
    const float nm = fmaxf(m_run, cmax);
    const float alpha = __expf(m_run - nm);
    float psum = 0.0f;
    bf16* sPp = &sP[w][0];
#pragma unroll
    for (int fm = 0; fm < 4; ++fm)
#pragma unroll
      for (int r = 0; r < 8; ++r) {
        const float pe = __expf(pv[fm][r] - nm);
        psum += pe;
        sPp[lq * 72 + fm * 16 + r + hi16 * 8] = f2bf(pe);
      }
    m_run = nm;
    l_run = l_run * alpha + psum;
#pragma unroll
    for (int fm = 0; fm < 4; ++fm)
#pragma unroll
      for (int r = 0; r < 8; ++r) cacc[fm].e[r] *= alpha;

    __syncthreads();  // order cross-lane sP writes before B-frag reads

    // ctx^T += V^T * P^T
#pragma unroll
    for (int st = 0; st < 2; ++st) {
      Frag16 bp;
      const bf16* pb = &sPp[lq * 72 + st * 32 + (hi16 ? 8 : 0)];
      bp.q[0] = *(const uint4*)pb;
      bp.q[1] = *(const uint4*)(pb + 16);
      const int koff = st * 32 + (hi16 ? 8 : 0);
#pragma unroll
      for (int fm = 0; fm < 4; ++fm) {
        Frag16 a;
        const bf16* p = &sVT[(fm * 16 + lq) * 72 + koff];
        a.q[0] = *(const uint4*)p;
        a.q[1] = *(const uint4*)(p + 16);
        cacc[fm].v = __builtin_amdgcn_wmma_f32_16x16x32_bf16(
            false, a.v, false, bp.v, (short)0, cacc[fm].v, false, false);
      }
    }
  }

  const float l_tot = l_run + __shfl_xor(l_run, 16);
  const float inv = 1.0f / l_tot;
  bf16* cp = ctx + ((long long)b * T_ + qrow) * (long long)(H_ * 64) + h * 64;
#pragma unroll
  for (int fm = 0; fm < 4; ++fm)
#pragma unroll
    for (int r = 0; r < 8; ++r)
      cp[fm * 16 + r + hi16 * 8] = f2bf(cacc[fm].e[r] * inv);
}

// ---------------------------------------------------------------------------
// Elementwise / norm kernels
// ---------------------------------------------------------------------------
__global__ void cast_f32_bf16(const float* __restrict__ s, bf16* __restrict__ d, long long n) {
  long long i = (long long)blockIdx.x * 256 + threadIdx.x;
  if (i < n) d[i] = f2bf(s[i]);
}

__global__ __launch_bounds__(256) void rmsnorm_bf16(const float* __restrict__ x,
                                                    const float* __restrict__ w,
                                                    bf16* __restrict__ o, int D_) {
  const long long row = blockIdx.x;
  const float* xr = x + row * D_;
  const int tid = threadIdx.x;
  float ss = 0.0f;
  for (int i = tid; i < D_; i += 256) { float v = xr[i]; ss += v * v; }
  for (int off = 16; off; off >>= 1) ss += __shfl_xor(ss, off);
  __shared__ float red[8];
  if ((tid & 31) == 0) red[tid >> 5] = ss;
  __syncthreads();
  float tot = 0.0f;
#pragma unroll
  for (int i = 0; i < 8; ++i) tot += red[i];
  const float sc = rsqrtf(tot / (float)D_ + 1e-5f);
  for (int i = tid; i < D_; i += 256) o[row * D_ + i] = f2bf(xr[i] * sc * w[i]);
}

__global__ void rope_f32_bf16(const float* __restrict__ src, bf16* __restrict__ dst,
                              int T_, long long total) {
  long long idx = (long long)blockIdx.x * 256 + threadIdx.x;
  if (idx >= total) return;
  const int i = (int)(idx & 31);
  const long long bht = idx >> 5;
  const int t = (int)(bht % T_);
  const float* r = src + bht * 64;
  const float invf = __powf(10000.0f, -(float)i * (1.0f / 32.0f));
  float sn, cs;
  __sincosf((float)t * invf, &sn, &cs);
  const float x1 = r[i], x2 = r[i + 32];
  dst[bht * 64 + i] = f2bf(x1 * cs - x2 * sn);
  dst[bht * 64 + i + 32] = f2bf(x2 * cs + x1 * sn);
}

__global__ void cast_transpose_v(const float* __restrict__ v, bf16* __restrict__ vt,
                                 int T_, long long total) {
  long long idx = (long long)blockIdx.x * 256 + threadIdx.x;
  if (idx >= total) return;
  const int dh = (int)(idx & 63);
  const long long rest = idx >> 6;
  const int t = (int)(rest % T_);
  const long long bk = rest / T_;
  vt[bk * 64LL * T_ + (long long)dh * T_ + t] = f2bf(v[idx]);
}

__global__ void silu_mul_bf16(const bf16* __restrict__ g, const bf16* __restrict__ u,
                              bf16* __restrict__ o, long long n) {
  long long i = (long long)blockIdx.x * 256 + threadIdx.x;
  if (i >= n) return;
  const float gv = bf2f(g[i]);
  const float s = gv / (1.0f + __expf(-gv));
  o[i] = f2bf(s * bf2f(u[i]));
}

// ---------------------------------------------------------------------------
// Launch
// ---------------------------------------------------------------------------
extern "C" void kernel_launch(void* const* d_in, const int* in_sizes, int n_in,
                              void* d_out, int out_size, void* d_ws, size_t ws_size,
                              hipStream_t stream) {
  (void)in_sizes; (void)n_in; (void)out_size; (void)ws_size;
  const int B = 4, T = 1024, D = 2048, H = 32, DH = 64, HK = 8, NI = 5632;
  const int BT = B * T;

  const float* x   = (const float*)d_in[0];
  const float* ln1 = (const float*)d_in[1];
  const float* ln2 = (const float*)d_in[2];

  char* base = (char*)d_ws;
  size_t off = 0;
  auto alloc = [&](size_t bytes) -> void* {
    off = (off + 255) & ~(size_t)255;
    void* r = base + off;
    off += bytes;
    return r;
  };

  // bf16 weights
  bf16* Wqkv = (bf16*)alloc((size_t)1536 * 2048 * 2);
  bf16* UsQ  = (bf16*)alloc((size_t)32 * 64 * 32 * 2);
  bf16* UsK  = (bf16*)alloc((size_t)8 * 64 * 32 * 2);
  bf16* UsV  = (bf16*)alloc((size_t)8 * 64 * 32 * 2);
  bf16* WoV  = (bf16*)alloc((size_t)1024 * 2048 * 2);
  bf16* WoUs = (bf16*)alloc((size_t)2048 * 1024 * 2);
  bf16* WgV  = (bf16*)alloc((size_t)1024 * 2048 * 2);
  bf16* WuV  = (bf16*)alloc((size_t)1024 * 2048 * 2);
  bf16* WgUs = (bf16*)alloc((size_t)NI * 1024 * 2);
  bf16* WuUs = (bf16*)alloc((size_t)NI * 1024 * 2);
  bf16* WdV  = (bf16*)alloc((size_t)1024 * NI * 2);
  bf16* WdUs = (bf16*)alloc((size_t)2048 * 1024 * 2);
  // activations
  bf16*  h1   = (bf16*)alloc((size_t)BT * D * 2);
  bf16*  rqkv = (bf16*)alloc((size_t)BT * 1536 * 2);
  float* qf   = (float*)alloc((size_t)B * H * T * DH * 4);
  float* kf   = (float*)alloc((size_t)B * HK * T * DH * 4);
  float* vf   = (float*)alloc((size_t)B * HK * T * DH * 4);
  bf16*  qrB  = (bf16*)alloc((size_t)B * H * T * DH * 2);
  bf16*  krB  = (bf16*)alloc((size_t)B * HK * T * DH * 2);
  bf16*  vtB  = (bf16*)alloc((size_t)B * HK * T * DH * 2);
  bf16*  ctxB = (bf16*)alloc((size_t)BT * D * 2);
  bf16*  orB  = (bf16*)alloc((size_t)BT * 1024 * 2);
  float* x2   = (float*)alloc((size_t)BT * D * 4);
  bf16*  h2   = (bf16*)alloc((size_t)BT * D * 2);
  bf16*  grB  = (bf16*)alloc((size_t)BT * 1024 * 2);
  bf16*  urB  = (bf16*)alloc((size_t)BT * 1024 * 2);
  bf16*  gB   = (bf16*)alloc((size_t)BT * NI * 2);
  bf16*  uB   = (bf16*)alloc((size_t)BT * NI * 2);
  bf16*  ffB  = (bf16*)alloc((size_t)BT * NI * 2);
  bf16*  drB  = (bf16*)alloc((size_t)BT * 1024 * 2);

  auto castN = [&](int idx, bf16* dst, long long n) {
    cast_f32_bf16<<<(unsigned)((n + 255) / 256), 256, 0, stream>>>((const float*)d_in[idx], dst, n);
  };
  castN(4,  Wqkv,                 1024LL * 2048);   // q_V  -> rows 0..1023
  castN(6,  Wqkv + 1024LL * 2048, 256LL  * 2048);   // k_V  -> rows 1024..1279
  castN(8,  Wqkv + 1280LL * 2048, 256LL  * 2048);   // v_V  -> rows 1280..1535
  castN(3,  UsQ,  32LL * 64 * 32);
  castN(5,  UsK,  8LL  * 64 * 32);
  castN(7,  UsV,  8LL  * 64 * 32);
  castN(10, WoV,  1024LL * 2048);
  castN(9,  WoUs, 2048LL * 1024);
  castN(12, WgV,  1024LL * 2048);
  castN(14, WuV,  1024LL * 2048);
  castN(11, WgUs, (long long)NI * 1024);
  castN(13, WuUs, (long long)NI * 1024);
  castN(16, WdV,  1024LL * NI);
  castN(15, WdUs, 2048LL * 1024);

  // h1 = rms(x) * ln1_w
  rmsnorm_bf16<<<BT, 256, 0, stream>>>(x, ln1, h1, D);

  // r_qkv[4096,1536] = h1 @ [q_V;k_V;v_V]^T
  launch_gemm<4, 2, 2, 4, 1>(stream, h1, Wqkv, rqkv, nullptr, BT, 1536, D, D, D, 1536);

  // per-head Us: batched [T,32] x [32,64]
  launch_gemm<2, 2, 2, 2, 0>(stream, rqkv, UsQ, qf, nullptr, T, 64, 32, 1536, 32, 64,
                             B * H, (long long)T * 1536, 32, 0, 64 * 32,
                             (long long)H * T * 64, (long long)T * 64, H);
  launch_gemm<2, 2, 2, 2, 0>(stream, rqkv + 1024, UsK, kf, nullptr, T, 64, 32, 1536, 32, 64,
                             B * HK, (long long)T * 1536, 32, 0, 64 * 32,
                             (long long)HK * T * 64, (long long)T * 64, HK);
  launch_gemm<2, 2, 2, 2, 0>(stream, rqkv + 1280, UsV, vf, nullptr, T, 64, 32, 1536, 32, 64,
                             B * HK, (long long)T * 1536, 32, 0, 64 * 32,
                             (long long)HK * T * 64, (long long)T * 64, HK);

  const long long nq = (long long)B * H * T * 32;
  rope_f32_bf16<<<(unsigned)((nq + 255) / 256), 256, 0, stream>>>(qf, qrB, T, nq);
  const long long nk = (long long)B * HK * T * 32;
  rope_f32_bf16<<<(unsigned)((nk + 255) / 256), 256, 0, stream>>>(kf, krB, T, nk);
  const long long nv = (long long)B * HK * T * 64;
  cast_transpose_v<<<(unsigned)((nv + 255) / 256), 256, 0, stream>>>(vf, vtB, T, nv);

  // attention -> ctx [B,T,H*DH] bf16
  attn_flash_wmma<<<dim3(T / 64, B * H), 128, 0, stream>>>(qrB, krB, vtB, ctxB, H, HK, T);

  // output projection + residual
  launch_gemm<4, 2, 2, 4, 1>(stream, ctxB, WoV, orB, nullptr, BT, 1024, 2048, 2048, 2048, 1024);
  launch_gemm<4, 2, 2, 4, 2>(stream, orB, WoUs, x2, x, BT, 2048, 1024, 1024, 1024, 2048);

  // MLP
  rmsnorm_bf16<<<BT, 256, 0, stream>>>(x2, ln2, h2, D);
  launch_gemm<4, 2, 2, 4, 1>(stream, h2, WgV, grB, nullptr, BT, 1024, 2048, 2048, 2048, 1024);
  launch_gemm<4, 2, 2, 4, 1>(stream, h2, WuV, urB, nullptr, BT, 1024, 2048, 2048, 2048, 1024);
  launch_gemm<4, 2, 2, 4, 1>(stream, grB, WgUs, gB, nullptr, BT, NI, 1024, 1024, 1024, NI);
  launch_gemm<4, 2, 2, 4, 1>(stream, urB, WuUs, uB, nullptr, BT, NI, 1024, 1024, 1024, NI);

  const long long nff = (long long)BT * NI;
  silu_mul_bf16<<<(unsigned)((nff + 255) / 256), 256, 0, stream>>>(gB, uB, ffB, nff);

  launch_gemm<4, 2, 2, 4, 1>(stream, ffB, WdV, drB, nullptr, BT, 1024, NI, NI, NI, 1024);
  launch_gemm<4, 2, 2, 4, 2>(stream, drB, WdUs, (float*)d_out, x2, BT, 2048, 1024, 1024, 1024, 2048);
}
